// EncodingLayer_13048110645276
// MI455X (gfx1250) — matrix operand order repaired and checked by reading, not compiled, not run on hardware
//
#include <hip/hip_runtime.h>

typedef __attribute__((ext_vector_type(16))) _Float16 v16h;
typedef __attribute__((ext_vector_type(8)))  float    v8f;

#define B_   8
#define L_   2048
#define D_   256
#define CDIM 512   // 2*D
#define NT_  (L_ / 32)   // 64 j-tiles

union Frag16 { unsigned int u[8]; v16h h; };

// Dynamic LDS, indexed by integer offsets only (keeps addrspace(3) provenance
// so fragment reads lower to ds_load_* instead of flat_load_*).
extern __shared__ _Float16 smemh[];

// A-fragment K index for dword v (low half h=2v), 16-bit 16x32 matrix
__device__ __forceinline__ int kpatA(int v, int lhalf) {
  return (lhalf ? 16 : 0) + 2 * v;
}

#define WMMA_F16(A, Bm, C) \
  __builtin_amdgcn_wmma_f32_16x16x32_f16(false, (A), false, (Bm), (short)0, (C), false, false)

// Async global->LDS copy, 16B per lane (512B per wave-instruction). ASYNCcnt.
__device__ __forceinline__ void async_copy_b128(unsigned lds_addr,
                                                const void* gbase,
                                                unsigned goff) {
  asm volatile("global_load_async_to_lds_b128 %0, %1, %2 offset:0"
               :: "v"(lds_addr), "v"(goff), "s"(gbase)
               : "memory");
}

template <int N>
__device__ __forceinline__ void s_wait_async() {
  asm volatile("s_wait_asynccnt %0" :: "i"(N) : "memory");
}

// ---------------------------------------------------------------------------
// Prep: f16 copies of x (into c[:, :256]) and q = x*wp
// ---------------------------------------------------------------------------
__global__ void prep_x_kernel(const float* __restrict__ x,
                              const float* __restrict__ w_itr_att,
                              _Float16* __restrict__ ch,
                              _Float16* __restrict__ qh) {
  int t = blockIdx.x * 256 + threadIdx.x;     // < B*L*D
  int row = t >> 8;
  int d = t & 255;
  float xv = x[t];
  ch[(size_t)row * CDIM + d] = (_Float16)xv;
  qh[t] = (_Float16)(xv * w_itr_att[2 * D_ + d]);  // wp = w_itr_att[2D:3D]
}

// Tiled transpose: xhT[b][d][j] = f16(x[b][j][d])  (d-major copy for P@V tiles)
__global__ void transpose_x_kernel(const float* __restrict__ x,
                                   _Float16* __restrict__ xhT) {
  __shared__ float tile[32][33];
  int blk = blockIdx.x;
  int b = blk / ((D_ / 32) * (L_ / 32));
  int rem = blk % ((D_ / 32) * (L_ / 32));
  int d0 = (rem / (L_ / 32)) * 32;
  int j0 = (rem % (L_ / 32)) * 32;
  int c = threadIdx.x & 31;
  int r = threadIdx.x >> 5;    // 0..7
  for (int rr = r; rr < 32; rr += 8)
    tile[rr][c] = x[((size_t)(b * L_ + j0 + rr)) * D_ + d0 + c];
  __syncthreads();
  for (int rr = r; rr < 32; rr += 8)
    xhT[((size_t)(b * D_ + d0 + rr)) * L_ + j0 + c] = (_Float16)tile[c][rr];
}

// s_u[j] = x[b,j,:] . wu  (s_m is constant along softmax axis -> dropped)
__global__ void su_kernel(const float* __restrict__ x,
                          const float* __restrict__ w_itr_att,
                          float* __restrict__ su) {
  int lane = threadIdx.x & 31;
  int row = blockIdx.x * 8 + (threadIdx.x >> 5);
  const float* xr = x + (size_t)row * D_;
  float acc = 0.0f;
  for (int d = lane; d < D_; d += 32) acc += xr[d] * w_itr_att[d];
  for (int off = 16; off > 0; off >>= 1) acc += __shfl_xor(acc, off, 32);
  if (lane == 0) su[row] = acc;
}

// Transposed f16 weights: whT[widx*256 + n][k] = w{1,2,3}[k][n]
__global__ void prep_w_kernel(const float* __restrict__ w1,
                              const float* __restrict__ w2,
                              const float* __restrict__ w3,
                              _Float16* __restrict__ whT) {
  int t = blockIdx.x * 256 + threadIdx.x;     // < 3*512*256
  int widx = t >> 17;
  int rem = t & 131071;
  int k = rem >> 8;
  int n = rem & 255;
  const float* w = (widx == 0) ? w1 : (widx == 1) ? w2 : w3;
  whT[((size_t)(widx * 256 + n)) * CDIM + k] = (_Float16)w[k * D_ + n];
}

// ---------------------------------------------------------------------------
// Flash-style attention with async double-buffered K/V staging.
// LDS (halfs): kv [2][32][264] @0, kvT [2][256][40] @16896, pst [4][16][36].
// Dynamic LDS total: 79360 bytes.
// ---------------------------------------------------------------------------
__global__ __launch_bounds__(128, 1)
void attn_kernel(const _Float16* __restrict__ qh,
                 _Float16* __restrict__ ch,
                 const _Float16* __restrict__ xhT,
                 const float* __restrict__ su) {
  constexpr int KV_SZ   = 32 * 264;
  constexpr int KVT_OFF = 2 * KV_SZ;
  constexpr int KVT_SZ  = 256 * 40;
  constexpr int PST_OFF = KVT_OFF + 2 * KVT_SZ;

  const int tid = threadIdx.x;
  const int lane = tid & 31;
  const int wave = tid >> 5;
  const int l15 = lane & 15;
  const int lhalf = (lane >> 4) & 1;
  const int b = blockIdx.x >> 5;
  const int i0 = (blockIdx.x & 31) * 64 + wave * 16;
  const int kB0 = lhalf * 16;

  const void* chbase = (const void*)(ch + (size_t)b * L_ * CDIM);
  const void* xtbase = (const void*)(xhT + (size_t)b * D_ * L_);

  // each wave issues 16 async b128 ops per tile (8 kv rows + 8 kvT row-groups)
  auto issue_tile = [&](int j0, int bufi) {
    unsigned kvl  = (unsigned)(size_t)&smemh[bufi * KV_SZ];
    unsigned kvtl = (unsigned)(size_t)&smemh[KVT_OFF + bufi * KVT_SZ];
    #pragma unroll
    for (int it = 0; it < 8; ++it) {
      int r = wave * 8 + it;                     // kv row (j within tile)
      async_copy_b128(kvl + (unsigned)r * 528u + lane * 16,
                      chbase, (unsigned)(j0 + r) * 1024u + lane * 16);
    }
    #pragma unroll
    for (int it = 0; it < 8; ++it) {
      int d = wave * 64 + it * 8 + (lane >> 2);  // kvT row (= d), 4 lanes/row
      int seg = lane & 3;
      async_copy_b128(kvtl + (unsigned)d * 80u + seg * 16,
                      xtbase, (unsigned)d * 4096u + (unsigned)j0 * 2u + seg * 16);
    }
  };

  // Resident Q fragments: 8 chunks of K=32 over D=256
  Frag16 aq[8];
  {
    const _Float16* qrow = qh + ((size_t)(b * L_ + i0 + l15)) * D_;
    #pragma unroll
    for (int c8 = 0; c8 < 8; ++c8)
      #pragma unroll
      for (int v = 0; v < 8; ++v)
        aq[c8].u[v] = *(const unsigned int*)(qrow + c8 * 32 + kpatA(v, lhalf));
  }

  Frag16 onesf;
  #pragma unroll
  for (int i = 0; i < 16; ++i) onesf.h[i] = (_Float16)1.0f;

  v8f o[16];
  v8f lacc;
  #pragma unroll
  for (int n8 = 0; n8 < 16; ++n8)
    #pragma unroll
    for (int e = 0; e < 8; ++e) o[n8][e] = 0.0f;
  #pragma unroll
  for (int e = 0; e < 8; ++e) lacc[e] = 0.0f;

  issue_tile(0, 0);

  for (int jt = 0; jt < NT_; ++jt) {
    const int cur = jt & 1;
    const int j0 = jt * 32;
    if (jt + 1 < NT_) {
      issue_tile(j0 + 32, cur ^ 1);
      s_wait_async<16>();   // loads complete in order -> tile jt is in LDS
    } else {
      s_wait_async<0>();
    }
    __syncthreads();        // all 4 waves' contributions visible

    const int kvo  = cur * KV_SZ;              // integer offsets, not pointers
    const int kvto = KVT_OFF + cur * KVT_SZ;
    float su01[2];
    su01[0] = su[b * L_ + j0 + l15];
    su01[1] = su[b * L_ + j0 + 16 + l15];

    // S = q @ k^T + s_u[j]; exp -> f16 P staged to LDS
    #pragma unroll
    for (int sub = 0; sub < 2; ++sub) {
      v8f s;
      #pragma unroll
      for (int e = 0; e < 8; ++e) s[e] = su01[sub];
      #pragma unroll
      for (int c8 = 0; c8 < 8; ++c8) {
        Frag16 bk;   // B[k][n] = x[j0+16sub+n][32c8+k]
        #pragma unroll
        for (int v = 0; v < 8; ++v)
          bk.u[v] = *(const unsigned int*)&smemh[kvo + (sub * 16 + l15) * 264 +
                                                 c8 * 32 + kB0 + 2 * v];
        s = WMMA_F16(aq[c8].h, bk.h, s);
      }
      #pragma unroll
      for (int e = 0; e < 8; ++e) {
        float p = __expf(s[e]);   // scores O(+-8): safe without max-subtract
        smemh[PST_OFF + (wave * 16 + e + 8 * lhalf) * 36 + sub * 16 + l15] =
            (_Float16)p;
      }
    }

    // P back as A-fragment (intra-wave DS ops are in-order)
    Frag16 ap;
    #pragma unroll
    for (int v = 0; v < 8; ++v)
      ap.u[v] = *(const unsigned int*)&smemh[PST_OFF + (wave * 16 + l15) * 36 +
                                             kpatA(v, lhalf)];

    // row-sum via ones trick: lands in C-layout matching o[]
    lacc = WMMA_F16(ap.h, onesf.h, lacc);

    // O += P @ V   (B[k][n] = x[j0+k][16n8+n] = kvT row n)
    #pragma unroll
    for (int n8 = 0; n8 < 16; ++n8) {
      Frag16 bv;
      #pragma unroll
      for (int v = 0; v < 8; ++v)
        bv.u[v] = *(const unsigned int*)&smemh[kvto + (n8 * 16 + l15) * 40 +
                                               kB0 + 2 * v];
      o[n8] = WMMA_F16(ap.h, bv.h, o[n8]);
    }
    __syncthreads();        // readers done before buffer cur is re-filled
  }

  float invl[8];
  #pragma unroll
  for (int e = 0; e < 8; ++e) invl[e] = 1.0f / lacc[e];
  _Float16* dst = ch + ((size_t)(b * L_ + i0)) * CDIM + D_;
  #pragma unroll
  for (int n8 = 0; n8 < 16; ++n8)
    #pragma unroll
    for (int e = 0; e < 8; ++e)
      dst[(size_t)(e + 8 * lhalf) * CDIM + n8 * 16 + l15] =
          (_Float16)(o[n8][e] * invl[e]);
}

// ---------------------------------------------------------------------------
// Fused MLP with async double-buffered weight slabs in LDS.
// LDS (halfs): wbuf [2][48][528]. Dynamic LDS total: 101376 bytes.
// ---------------------------------------------------------------------------
__global__ __launch_bounds__(128, 1)
void mlp_kernel(const _Float16* __restrict__ ch,
                const _Float16* __restrict__ whT,   // [768][512]
                const float* __restrict__ x,
                const float* __restrict__ b1,
                const float* __restrict__ b2,
                const float* __restrict__ b3,
                float* __restrict__ out) {
  constexpr int WB_SZ = 48 * 528;

  const int tid = threadIdx.x;
  const int lane = tid & 31;
  const int wave = tid >> 5;
  const int l15 = lane & 15;
  const int lhalf = (lane >> 4) & 1;
  const int row0 = blockIdx.x * 64 + wave * 16;
  const int kB0 = lhalf * 16;

  // each wave: 12 rows x 2 parts = 24 async b128 ops per slab
  auto issue_slab = [&](int d0, int bufi) {
    unsigned wl = (unsigned)(size_t)&smemh[bufi * WB_SZ];
    #pragma unroll
    for (int it = 0; it < 12; ++it) {
      int r = wave * 12 + it;            // 0..47
      int widx = r >> 4, r15 = r & 15;
      unsigned grow = (unsigned)(widx * 256 + d0 + r15) * 1024u;
      #pragma unroll
      for (int part = 0; part < 2; ++part)
        async_copy_b128(wl + (unsigned)r * 1056u + part * 512u + lane * 16,
                        (const void*)whT, grow + part * 512u + lane * 16);
    }
  };

  // resident c-row A-fragments: 16 chunks of K=32 over 512
  Frag16 ac[16];
  {
    const _Float16* crow = ch + (size_t)(row0 + l15) * CDIM;
    #pragma unroll
    for (int kc = 0; kc < 16; ++kc)
      #pragma unroll
      for (int v = 0; v < 8; ++v)
        ac[kc].u[v] = *(const unsigned int*)(crow + kc * 32 + kpatA(v, lhalf));
  }

  issue_slab(0, 0);

  for (int d8 = 0; d8 < 16; ++d8) {
    const int cur = d8 & 1;
    if (d8 + 1 < 16) {
      issue_slab((d8 + 1) * 16, cur ^ 1);
      s_wait_async<24>();
    } else {
      s_wait_async<0>();
    }
    __syncthreads();

    const int wo = cur * WB_SZ;          // integer offset select
    const int col = d8 * 16 + l15;
    v8f za, ra, fa;
    float vb1 = b1[col], vb2 = b2[col], vb3 = b3[col];
    #pragma unroll
    for (int e = 0; e < 8; ++e) { za[e] = vb1; ra[e] = vb2; fa[e] = vb3; }

    #pragma unroll
    for (int kc = 0; kc < 16; ++kc) {
      Frag16 bz, br, bf;
      #pragma unroll
      for (int v = 0; v < 8; ++v) {
        int k = kc * 32 + kB0 + 2 * v;
        bz.u[v] = *(const unsigned int*)&smemh[wo + (0  + l15) * 528 + k];
        br.u[v] = *(const unsigned int*)&smemh[wo + (16 + l15) * 528 + k];
        bf.u[v] = *(const unsigned int*)&smemh[wo + (32 + l15) * 528 + k];
      }
      za = WMMA_F16(ac[kc].h, bz.h, za);
      ra = WMMA_F16(ac[kc].h, br.h, ra);
      fa = WMMA_F16(ac[kc].h, bf.h, fa);
    }

    #pragma unroll
    for (int e = 0; e < 8; ++e) {
      int row = row0 + e + 8 * lhalf;
      float xv = x[(size_t)row * D_ + col];
      float zz = tanhf(za[e]);
      float rr = 1.0f / (1.0f + __expf(-ra[e]));
      float ff = 1.0f / (1.0f + __expf(-fa[e]));
      out[(size_t)row * D_ + col] = rr * xv + ff * zz;
    }
    __syncthreads();
  }
}

// ---------------------------------------------------------------------------
extern "C" void kernel_launch(void* const* d_in, const int* in_sizes, int n_in,
                              void* d_out, int out_size, void* d_ws, size_t ws_size,
                              hipStream_t stream) {
  (void)in_sizes; (void)n_in; (void)out_size; (void)ws_size;
  const float* x  = (const float*)d_in[0];
  const float* wi = (const float*)d_in[1];
  const float* w1 = (const float*)d_in[2];
  const float* w2 = (const float*)d_in[3];
  const float* w3 = (const float*)d_in[4];
  const float* b1 = (const float*)d_in[5];
  const float* b2 = (const float*)d_in[6];
  const float* b3 = (const float*)d_in[7];
  float* out = (float*)d_out;

  const size_t N = (size_t)B_ * L_;            // 16384 rows
  char* ws = (char*)d_ws;
  _Float16* ch  = (_Float16*)(ws);                         // 16 MB [N][512]
  _Float16* qh  = (_Float16*)(ws + 16777216);              // 8 MB  [N][256]
  _Float16* xhT = (_Float16*)(ws + 25165824);              // 8 MB  [B][256][2048]
  float*    su  = (float*)   (ws + 33554432);              // 64 KB
  _Float16* whT = (_Float16*)(ws + 33619968);              // 768 KB [768][512]

  prep_x_kernel<<<(unsigned)((N * D_) / 256), 256, 0, stream>>>(x, wi, ch, qh);
  transpose_x_kernel<<<B_ * (D_ / 32) * (L_ / 32), 256, 0, stream>>>(x, xhT);
  su_kernel<<<(unsigned)(N / 8), 256, 0, stream>>>(x, wi, su);
  prep_w_kernel<<<1536, 256, 0, stream>>>(w1, w2, w3, whT);
  attn_kernel<<<B_ * (L_ / 64), 128, 79360, stream>>>(qh, ch, xhT, su);
  mlp_kernel<<<(unsigned)(N / 64), 128, 101376, stream>>>(ch, whT, x, b1, b2, b3, out);
}
